// Recommender_52209622450607
// MI455X (gfx1250) — matrix operand order repaired and checked by reading, not compiled, not run on hardware
//
#include <hip/hip_runtime.h>
#include <hip/hip_bf16.h>
#include <math.h>

#define GNN_N   50000
#define GNN_E   800000
#define GNN_IN  128
#define GNN_OUT 64
#define GNN_H   4
#define GNN_HC  (GNN_H * GNN_OUT)   // 256
#define GNN_EPS 1e-16f

typedef __attribute__((ext_vector_type(16))) _Float16 v16h;
typedef __attribute__((ext_vector_type(8)))  _Float16 v8h;
typedef __attribute__((ext_vector_type(4)))  _Float16 v4h;
typedef __attribute__((ext_vector_type(8)))  float    v8f;

// ---------------------------------------------------------------------------
// f32 -> f16 bulk convert (count4 = #float4 chunks; sizes here are mult. of 4)
// ---------------------------------------------------------------------------
__global__ void cvt_f16_kernel(const float* __restrict__ x,
                               _Float16* __restrict__ out, int count4) {
    int i = blockIdx.x * blockDim.x + threadIdx.x;
    if (i >= count4) return;
    float4 v = ((const float4*)x)[i];
    v4h o;
    o[0] = (_Float16)v.x; o[1] = (_Float16)v.y;
    o[2] = (_Float16)v.z; o[3] = (_Float16)v.w;
    ((v4h*)out)[i] = o;
}

// ---------------------------------------------------------------------------
// Pack W[K,Nc] (f32, row-major) into WMMA-B lane layout, f16:
// group g = (tileN*ksteps + kidx)*32 + lane holds 16 contiguous f16 =
// column (tileN*16 + lane%16), K rows kidx*32 + (lane/16)*16 + e.
// One thread per group; runs once, then GEMM B-loads are pure b128s.
// ---------------------------------------------------------------------------
__global__ void pack_w_kernel(const float* __restrict__ W,
                              _Float16* __restrict__ Wp, int K, int Nc) {
    const int ksteps = K >> 5;
    const int tilesN = Nc >> 4;
    int t = blockIdx.x * blockDim.x + threadIdx.x;
    if (t >= tilesN * ksteps * 32) return;
    const int lane = t & 31;
    const int kidx = (t >> 5) % ksteps;
    const int tile = (t >> 5) / ksteps;
    const int col  = (tile << 4) + (lane & 15);
    const int kb   = (kidx << 5) + ((lane >> 4) << 4);
    _Float16* out = Wp + ((size_t)t << 4);
#pragma unroll
    for (int e = 0; e < 16; ++e)
        out[e] = (_Float16)W[(size_t)(kb + e) * Nc + col];
}

// ---------------------------------------------------------------------------
// WMMA GEMM:  Y[M,Nc] = Xh[M,K](f16) @ W + bias, W pre-packed (Wp).
// One wave32 owns a 16x64 output strip (4 accumulators); per k-step:
// 2 b128 A loads (reused 4x) + 8 b128 B loads -> 4 v_wmma_f32_16x16x32_f16.
// K mult. of 32, Nc mult. of 64 (here K in {128,256}, Nc in {256,64}).
// ---------------------------------------------------------------------------
__global__ void gemm_wmma_f16_kernel(const _Float16* __restrict__ Xh,
                                     const _Float16* __restrict__ Wp,
                                     const float* __restrict__ bias,
                                     float* __restrict__ Y,
                                     int M, int K, int Nc) {
    const int lane       = threadIdx.x & 31;
    const int wavesPerBk = blockDim.x >> 5;
    const int wid        = blockIdx.x * wavesPerBk + (threadIdx.x >> 5);
    const int tilesN4    = Nc >> 6;                 // 64 output cols per wave
    const int tilesM     = (M + 15) >> 4;
    if (wid >= tilesM * tilesN4) return;            // uniform per wave
    const int tm   = wid / tilesN4;
    const int tn4  = wid - tm * tilesN4;
    const int row0 = tm << 4;
    const int half = lane >> 4;                     // A: K 0-15 vs 16-31
    const int l16  = lane & 15;
    const int ksteps = K >> 5;

    int arow = row0 + l16;
    if (arow >= M) arow = M - 1;                    // clamp: no EXEC divergence
    const v8h* xp = (const v8h*)(Xh + (size_t)arow * K + (half << 4));
    // B group base (v8h units): ((tile*ksteps + kidx)*32 + lane)*2
    const v8h* wp = (const v8h*)Wp;
    const size_t btile0 = ((size_t)(tn4 << 2) * ksteps) * 32;

    v8f acc[4] = {};
    for (int kidx = 0; kidx < ksteps; ++kidx) {
        v8h alo = xp[kidx * 4];                     // k0+half*16 .. +7
        v8h ahi = xp[kidx * 4 + 1];                 // k0+half*16+8 .. +7
        v16h a = __builtin_shufflevector(alo, ahi,
                 0,1,2,3,4,5,6,7,8,9,10,11,12,13,14,15);
#pragma unroll
        for (int t = 0; t < 4; ++t) {
            const v8h* bpt = wp +
                ((btile0 + ((size_t)t * ksteps + kidx) * 32 + lane) << 1);
            v16h b = __builtin_shufflevector(bpt[0], bpt[1],
                     0,1,2,3,4,5,6,7,8,9,10,11,12,13,14,15);
            acc[t] = __builtin_amdgcn_wmma_f32_16x16x32_f16(
                false, a, false, b, (short)0, acc[t], false, false);
        }
    }

#pragma unroll
    for (int t = 0; t < 4; ++t) {
        const int col = (tn4 << 6) + (t << 4) + l16;
        const float bv = bias[col];
#pragma unroll
        for (int r = 0; r < 8; ++r) {
            const int orow = row0 + r + (half << 3); // D: VGPR r -> row r / r+8
            if (orow < M) Y[(size_t)orow * Nc + col] = acc[t][r] + bv;
        }
    }
}

// ---------------------------------------------------------------------------
// Edge-phase kernels (gather + atomic bound; no matrix shape)
// ---------------------------------------------------------------------------
__global__ void fill_kernel(float* __restrict__ p, float v, int count) {
    int i = blockIdx.x * blockDim.x + threadIdx.x;
    if (i < count) p[i] = v;
}

__device__ __forceinline__ void atomicMaxF(float* addr, float val) {
    // IEEE-ordering bit trick; buffer pre-initialized to -inf.
    if (val >= 0.0f) atomicMax((int*)addr, __float_as_int(val));
    else             atomicMin((unsigned int*)addr, (unsigned int)__float_as_int(val));
}

// alpha[e,h] = <q[dst,h,:], k[src,h,:]> / sqrt(C);  running segment max
__global__ void edge_alpha_kernel(const float* __restrict__ q,
                                  const float* __restrict__ k,
                                  const int* __restrict__ src,
                                  const int* __restrict__ dst,
                                  float* __restrict__ alpha,
                                  float* __restrict__ amax,
                                  int E, int heads, int C) {
    int e = blockIdx.x * blockDim.x + threadIdx.x;
    if (e >= E) return;
    const int s = src[e], d = dst[e];
    const float scale = rsqrtf((float)C);
    const float* qp = q + (size_t)d * heads * C;
    const float* kp = k + (size_t)s * heads * C;
    for (int h = 0; h < heads; ++h) {
        const float4* q4 = (const float4*)(qp + h * C);
        const float4* k4 = (const float4*)(kp + h * C);
        float accd = 0.0f;
        for (int c = 0; c < (C >> 2); ++c) {
            float4 qq = q4[c], kk = k4[c];
            accd += qq.x * kk.x + qq.y * kk.y + qq.z * kk.z + qq.w * kk.w;
        }
        accd *= scale;
        alpha[(size_t)e * heads + h] = accd;
        atomicMaxF(&amax[(size_t)d * heads + h], accd);
    }
}

// ex = exp(alpha - amax[dst]);  alpha <- ex;  denom[dst] += ex
__global__ void edge_exp_kernel(float* __restrict__ alpha,
                                const int* __restrict__ dst,
                                const float* __restrict__ amax,
                                float* __restrict__ denom,
                                int E, int heads) {
    int t = blockIdx.x * blockDim.x + threadIdx.x;
    if (t >= E * heads) return;
    const int e = t / heads, h = t - e * heads;
    const int d = dst[e];
    const float ex = __expf(alpha[t] - amax[(size_t)d * heads + h]);
    alpha[t] = ex;
    atomicAdd(&denom[(size_t)d * heads + h], ex);
}

// agg[dst,h,:] += v[src,h,:] * ex/(denom+eps); thread per (edge, head, 4ch)
__global__ void edge_agg_kernel(const float* __restrict__ v,
                                const float* __restrict__ ex,
                                const float* __restrict__ denom,
                                const int* __restrict__ src,
                                const int* __restrict__ dst,
                                float* __restrict__ agg,
                                int E, int heads, int C) {
    const int chunks = heads * (C >> 2);
    long long t = (long long)blockIdx.x * blockDim.x + threadIdx.x;
    if (t >= (long long)E * chunks) return;
    const int e   = (int)(t / chunks);
    const int rem = (int)(t - (long long)e * chunks);
    const int h   = rem / (C >> 2);
    const int c4  = rem - h * (C >> 2);
    const int s = src[e], d = dst[e];
    const float attn = ex[(size_t)e * heads + h] /
                       (denom[(size_t)d * heads + h] + GNN_EPS);
    const float4 vv = ((const float4*)(v + ((size_t)s * heads + h) * C))[c4];
    float* ap = agg + ((size_t)d * heads + h) * C + c4 * 4;
    atomicAdd(ap + 0, vv.x * attn);
    atomicAdd(ap + 1, vv.y * attn);
    atomicAdd(ap + 2, vv.z * attn);
    atomicAdd(ap + 3, vv.w * attn);
}

// Gated beta skip: one wave32 per node, wave-reduce the gate logit.
__global__ void beta_kernel(const float* __restrict__ agg,
                            const float* __restrict__ skip,
                            const float* __restrict__ wb,
                            float* __restrict__ out,
                            int n, int F, int do_relu) {
    const int node = blockIdx.x * (blockDim.x >> 5) + (threadIdx.x >> 5);
    if (node >= n) return;
    const int lane = threadIdx.x & 31;
    const float* op = agg  + (size_t)node * F;
    const float* rp = skip + (size_t)node * F;
    float g = 0.0f;
    for (int c = lane; c < F; c += 32) {
        const float o = op[c], r = rp[c];
        g += o * wb[c] + r * wb[F + c] + (o - r) * wb[2 * F + c];
    }
#pragma unroll
    for (int off = 16; off >= 1; off >>= 1) g += __shfl_xor(g, off, 32);
    const float beta = 1.0f / (1.0f + __expf(-g));
    for (int c = lane; c < F; c += 32) {
        const float o = op[c], r = rp[c];
        float y = beta * r + (1.0f - beta) * o;
        if (do_relu) y = fmaxf(y, 0.0f);
        out[(size_t)node * F + c] = y;
    }
}

// log_softmax over C channels + raw copy; one wave32 per node.
__global__ void logsoftmax_kernel(const float* __restrict__ x,
                                  float* __restrict__ out_ls,
                                  float* __restrict__ out_raw,
                                  int n, int C) {
    const int node = blockIdx.x * (blockDim.x >> 5) + (threadIdx.x >> 5);
    if (node >= n) return;
    const int lane = threadIdx.x & 31;
    const float* xp = x + (size_t)node * C;
    float m = -INFINITY;
    for (int c = lane; c < C; c += 32) m = fmaxf(m, xp[c]);
#pragma unroll
    for (int off = 16; off >= 1; off >>= 1) m = fmaxf(m, __shfl_xor(m, off, 32));
    float s = 0.0f;
    for (int c = lane; c < C; c += 32) s += __expf(xp[c] - m);
#pragma unroll
    for (int off = 16; off >= 1; off >>= 1) s += __shfl_xor(s, off, 32);
    const float lse = m + __logf(s);
    for (int c = lane; c < C; c += 32) {
        const float xv = xp[c];
        out_ls [(size_t)node * C + c] = xv - lse;
        out_raw[(size_t)node * C + c] = xv;
    }
}

// ---------------------------------------------------------------------------
// Orchestration
// ---------------------------------------------------------------------------
extern "C" void kernel_launch(void* const* d_in, const int* in_sizes, int n_in,
                              void* d_out, int out_size, void* d_ws, size_t ws_size,
                              hipStream_t stream) {
    (void)in_sizes; (void)n_in; (void)out_size; (void)ws_size;
    const int N = GNN_N, E = GNN_E, IN = GNN_IN, OUT = GNN_OUT, H = GNN_H, HC = GNN_HC;

    const float* x   = (const float*)d_in[0];
    const int*   ei  = (const int*)d_in[1];
    const int*   src = ei;
    const int*   dst = ei + E;
    // d_in[2] = edge_weight: ignored (edge_dim=None in reference)
    const float* wq1 = (const float*)d_in[3];  const float* bq1 = (const float*)d_in[4];
    const float* wk1 = (const float*)d_in[5];  const float* bk1 = (const float*)d_in[6];
    const float* wv1 = (const float*)d_in[7];  const float* bv1 = (const float*)d_in[8];
    const float* ws1 = (const float*)d_in[9];  const float* bs1 = (const float*)d_in[10];
    const float* wb1 = (const float*)d_in[11];
    const float* wq2 = (const float*)d_in[12]; const float* bq2 = (const float*)d_in[13];
    const float* wk2 = (const float*)d_in[14]; const float* bk2 = (const float*)d_in[15];
    const float* wv2 = (const float*)d_in[16]; const float* bv2 = (const float*)d_in[17];
    const float* ws2 = (const float*)d_in[18]; const float* bs2 = (const float*)d_in[19];
    const float* wb2 = (const float*)d_in[20];

    // Bump-allocate scratch out of d_ws (float units; all counts mult. of 4).
    float* ws = (float*)d_ws;
    size_t off = 0;
    auto alloc = [&](size_t cnt) { float* p = ws + off; off += cnt; return p; };
    float* q1   = alloc((size_t)N * HC);
    float* k1   = alloc((size_t)N * HC);
    float* v1   = alloc((size_t)N * HC);
    float* s1   = alloc((size_t)N * HC);
    float* agg1 = alloc((size_t)N * HC);
    float* h1   = alloc((size_t)N * HC);
    float* al1  = alloc((size_t)E * H);
    float* mx1  = alloc((size_t)N * H);
    float* dn1  = alloc((size_t)N * H);
    float* q2   = alloc((size_t)N * OUT);
    float* k2   = alloc((size_t)N * OUT);
    float* v2   = alloc((size_t)N * OUT);
    float* s2   = alloc((size_t)N * OUT);
    float* agg2 = alloc((size_t)N * OUT);
    float* o2   = alloc((size_t)N * OUT);
    float* al2  = alloc((size_t)E);
    float* mx2  = alloc((size_t)N);
    float* dn2  = alloc((size_t)N);
    // f16 buffers (element counts mult. of 8 -> float counts mult. of 4)
    _Float16* xh  = (_Float16*)alloc((size_t)N * IN / 2);
    _Float16* h1h = (_Float16*)alloc((size_t)N * HC / 2);
    _Float16* wp1q = (_Float16*)alloc((size_t)IN * HC / 2);
    _Float16* wp1k = (_Float16*)alloc((size_t)IN * HC / 2);
    _Float16* wp1v = (_Float16*)alloc((size_t)IN * HC / 2);
    _Float16* wp1s = (_Float16*)alloc((size_t)IN * HC / 2);
    _Float16* wp2q = (_Float16*)alloc((size_t)HC * OUT / 2);
    _Float16* wp2k = (_Float16*)alloc((size_t)HC * OUT / 2);
    _Float16* wp2v = (_Float16*)alloc((size_t)HC * OUT / 2);
    _Float16* wp2s = (_Float16*)alloc((size_t)HC * OUT / 2);

    const int TB = 256;                  // 8 wave32 per block
    auto cdiv = [](long long a, long long b) { return (int)((a + b - 1) / b); };

    // ---------------- Pre-pass: f16 convert + weight pack -------------------
    cvt_f16_kernel<<<cdiv((long long)N * IN / 4, TB), TB, 0, stream>>>(x, xh, N * IN / 4);
    {
        const int g1 = cdiv((IN / 32) * (HC / 16) * 32, TB);   // 2048 groups
        pack_w_kernel<<<g1, TB, 0, stream>>>(wq1, wp1q, IN, HC);
        pack_w_kernel<<<g1, TB, 0, stream>>>(wk1, wp1k, IN, HC);
        pack_w_kernel<<<g1, TB, 0, stream>>>(wv1, wp1v, IN, HC);
        pack_w_kernel<<<g1, TB, 0, stream>>>(ws1, wp1s, IN, HC);
        const int g2 = cdiv((HC / 32) * (OUT / 16) * 32, TB);  // 1024 groups
        pack_w_kernel<<<g2, TB, 0, stream>>>(wq2, wp2q, HC, OUT);
        pack_w_kernel<<<g2, TB, 0, stream>>>(wk2, wp2k, HC, OUT);
        pack_w_kernel<<<g2, TB, 0, stream>>>(wv2, wp2v, HC, OUT);
        pack_w_kernel<<<g2, TB, 0, stream>>>(ws2, wp2s, HC, OUT);
    }

    // ---------------- Layer 1: IN=128 -> HC=256, heads=4, concat ------------
    {
        const int waves = cdiv(N, 16) * (HC / 64);
        const int grid  = cdiv(waves, TB / 32);
        gemm_wmma_f16_kernel<<<grid, TB, 0, stream>>>(xh, wp1q, bq1, q1, N, IN, HC);
        gemm_wmma_f16_kernel<<<grid, TB, 0, stream>>>(xh, wp1k, bk1, k1, N, IN, HC);
        gemm_wmma_f16_kernel<<<grid, TB, 0, stream>>>(xh, wp1v, bv1, v1, N, IN, HC);
        gemm_wmma_f16_kernel<<<grid, TB, 0, stream>>>(xh, wp1s, bs1, s1, N, IN, HC);
    }
    fill_kernel<<<cdiv((long long)N * H, TB), TB, 0, stream>>>(mx1, -INFINITY, N * H);
    fill_kernel<<<cdiv((long long)N * H, TB), TB, 0, stream>>>(dn1, 0.0f, N * H);
    fill_kernel<<<cdiv((long long)N * HC, TB), TB, 0, stream>>>(agg1, 0.0f, N * HC);

    edge_alpha_kernel<<<cdiv(E, TB), TB, 0, stream>>>(q1, k1, src, dst, al1, mx1, E, H, OUT);
    edge_exp_kernel<<<cdiv((long long)E * H, TB), TB, 0, stream>>>(al1, dst, mx1, dn1, E, H);
    edge_agg_kernel<<<cdiv((long long)E * H * (OUT / 4), TB), TB, 0, stream>>>(
        v1, al1, dn1, src, dst, agg1, E, H, OUT);
    beta_kernel<<<cdiv(N, TB / 32), TB, 0, stream>>>(agg1, s1, wb1, h1, N, HC, /*relu=*/1);

    // ---------------- Layer 2: HC=256 -> OUT=64, heads=1 --------------------
    cvt_f16_kernel<<<cdiv((long long)N * HC / 4, TB), TB, 0, stream>>>(h1, h1h, N * HC / 4);
    {
        const int waves = cdiv(N, 16) * (OUT / 64);
        const int grid  = cdiv(waves, TB / 32);
        gemm_wmma_f16_kernel<<<grid, TB, 0, stream>>>(h1h, wp2q, bq2, q2, N, HC, OUT);
        gemm_wmma_f16_kernel<<<grid, TB, 0, stream>>>(h1h, wp2k, bk2, k2, N, HC, OUT);
        gemm_wmma_f16_kernel<<<grid, TB, 0, stream>>>(h1h, wp2v, bv2, v2, N, HC, OUT);
        gemm_wmma_f16_kernel<<<grid, TB, 0, stream>>>(h1h, wp2s, bs2, s2, N, HC, OUT);
    }
    fill_kernel<<<cdiv(N, TB), TB, 0, stream>>>(mx2, -INFINITY, N);
    fill_kernel<<<cdiv(N, TB), TB, 0, stream>>>(dn2, 0.0f, N);
    fill_kernel<<<cdiv((long long)N * OUT, TB), TB, 0, stream>>>(agg2, 0.0f, N * OUT);

    edge_alpha_kernel<<<cdiv(E, TB), TB, 0, stream>>>(q2, k2, src, dst, al2, mx2, E, 1, OUT);
    edge_exp_kernel<<<cdiv(E, TB), TB, 0, stream>>>(al2, dst, mx2, dn2, E, 1);
    edge_agg_kernel<<<cdiv((long long)E * (OUT / 4), TB), TB, 0, stream>>>(
        v2, al2, dn2, src, dst, agg2, E, 1, OUT);
    beta_kernel<<<cdiv(N, TB / 32), TB, 0, stream>>>(agg2, s2, wb2, o2, N, OUT, /*relu=*/0);

    // ---------------- Output: (log_softmax(out), out) -----------------------
    float* out_ls  = (float*)d_out;
    float* out_raw = (float*)d_out + (size_t)N * OUT;
    logsoftmax_kernel<<<cdiv(N, TB / 32), TB, 0, stream>>>(o2, out_ls, out_raw, N, OUT);
}
